// FastRGCNGNN_19439021982260
// MI455X (gfx1250) — matrix operand order repaired and checked by reading, not compiled, not run on hardware
//
#include <hip/hip_runtime.h>
#include <hip/hip_bf16.h>

#define NODES 100000
#define EDGES 1600000
#define IN_DIM 128
#define HID_DIM 32
#define OUT_DIM 64
#define N_REL 8

typedef __attribute__((ext_vector_type(16))) __bf16 v16bf;
typedef __attribute__((ext_vector_type(8)))  __bf16 v8bf;
typedef __attribute__((ext_vector_type(8)))  float  v8f;

// ---- WMMA fragment loaders --------------------------------------------------
// A fragment (16x32 bf16, row-major [M][K] bf16 source):
// lane L holds row M = L&15; element e holds K = kbase + e + (L>=16?8:0) + (e>=8?8:0)
// -> two contiguous 8-element (16 B) loads per lane.
__device__ __forceinline__ v16bf load_a_frag_bf(const __bf16* __restrict__ A, int lda,
                                                int row, int kbase, int lane) {
    const __bf16* p = A + (size_t)row * lda + kbase + ((lane & 16) ? 8 : 0);
    v8bf lo = *(const v8bf*)p;
    v8bf hi = *(const v8bf*)(p + 16);
    return __builtin_shufflevector(lo, hi, 0, 1, 2, 3, 4, 5, 6, 7,
                                   8, 9, 10, 11, 12, 13, 14, 15);
}

// A fragment from f32 source (for atomically-accumulated activations).
__device__ __forceinline__ v16bf load_a_frag_f32(const float* __restrict__ A, int lda,
                                                 int row, int kbase, int lane) {
    const float* p = A + (size_t)row * lda + kbase + ((lane & 16) ? 8 : 0);
    v16bf r;
#pragma unroll
    for (int e = 0; e < 8; ++e) r[e] = (__bf16)p[e];
#pragma unroll
    for (int e = 0; e < 8; ++e) r[8 + e] = (__bf16)p[16 + e];
    return r;
}

// B fragment (32x16) from bf16 TRANSPOSED weights BT[N][K] (ldk = K length):
// lane L holds column n; element e holds K = kbase + e + (L>=16?8:0) + (e>=8?8:0)
// -> two contiguous 8-element (16 B) loads per lane.
__device__ __forceinline__ v16bf load_b_frag_bf(const __bf16* __restrict__ BT, int ldk,
                                                int col, int kbase, int lane) {
    const __bf16* p = BT + (size_t)col * ldk + kbase + ((lane & 16) ? 8 : 0);
    v8bf lo = *(const v8bf*)p;
    v8bf hi = *(const v8bf*)(p + 16);
    return __builtin_shufflevector(lo, hi, 0, 1, 2, 3, 4, 5, 6, 7,
                                   8, 9, 10, 11, 12, 13, 14, 15);
}

// C/D f32 16x16: lane L -> column N=L&15; VGPR v -> row M = v + (L>=16?8:0)
__device__ __forceinline__ void store_tile(float* __restrict__ D, int ldd,
                                           int row0, int col0, int lane, v8f acc) {
    int n = lane & 15;
    int g = (lane & 16) ? 8 : 0;
#pragma unroll
    for (int v = 0; v < 8; ++v)
        D[(size_t)(row0 + g + v) * ldd + col0 + n] = acc[v];
}

__device__ __forceinline__ v8f wmma_bf16(v16bf a, v16bf b, v8f c) {
    return __builtin_amdgcn_wmma_f32_16x16x32_bf16(false, a, false, b, (short)0, c,
                                                   false, false);
}

// ---- Precision / layout conversion kernels ---------------------------------

__global__ void k_zero(float* __restrict__ p, long long count) {
    long long i = (long long)blockIdx.x * blockDim.x + threadIdx.x;
    long long stride = (long long)gridDim.x * blockDim.x;
    for (; i < count; i += stride) p[i] = 0.0f;
}

// x (f32 [N][128]) -> bf16, vectorized 8 elements/thread.
__global__ void k_cvt_x(const float* __restrict__ x, __bf16* __restrict__ xb) {
    long long i = ((long long)blockIdx.x * blockDim.x + threadIdx.x) * 8;
    if (i >= (long long)NODES * IN_DIM) return;
    v8bf o;
#pragma unroll
    for (int e = 0; e < 8; ++e) o[e] = (__bf16)x[i + e];
    *(v8bf*)(xb + i) = o;
}

// Weights -> bf16, transposed to [N][K] for contiguous B-fragment loads.
__global__ void k_cvt_weights(const float* __restrict__ W1, const float* __restrict__ root1,
                              const float* __restrict__ W2, const float* __restrict__ root2,
                              __bf16* __restrict__ W1T, __bf16* __restrict__ root1T,
                              __bf16* __restrict__ W2T, __bf16* __restrict__ root2T) {
    int i = blockIdx.x * blockDim.x + threadIdx.x;
    if (i < N_REL * IN_DIM * HID_DIM) {                       // W1 [R][128][32]
        int r = i / (IN_DIM * HID_DIM), rem = i % (IN_DIM * HID_DIM);
        int k = rem / HID_DIM, n = rem % HID_DIM;
        W1T[((size_t)r * HID_DIM + n) * IN_DIM + k] = (__bf16)W1[i];
    }
    if (i < IN_DIM * HID_DIM) {                               // root1 [128][32]
        int k = i / HID_DIM, n = i % HID_DIM;
        root1T[(size_t)n * IN_DIM + k] = (__bf16)root1[i];
    }
    if (i < N_REL * HID_DIM * OUT_DIM) {                      // W2 [R][32][64]
        int r = i / (HID_DIM * OUT_DIM), rem = i % (HID_DIM * OUT_DIM);
        int k = rem / OUT_DIM, n = rem % OUT_DIM;
        W2T[((size_t)r * OUT_DIM + n) * HID_DIM + k] = (__bf16)W2[i];
    }
    if (i < HID_DIM * OUT_DIM) {                              // root2 [32][64]
        int k = i / OUT_DIM, n = i % OUT_DIM;
        root2T[(size_t)n * HID_DIM + k] = (__bf16)root2[i];
    }
}

// ---- Layer 1 ----------------------------------------------------------------

// grid.y = 0..7 -> xW1[:,r,:] = x @ W1[r]; grid.y = 8 -> rootout1 = x @ root1.
// One wave per 16-node tile; HID=32 -> 2 column tiles, K=128 -> 4 WMMA steps each.
__global__ void k_transform1(const __bf16* __restrict__ xb,
                             const __bf16* __restrict__ W1T,
                             const __bf16* __restrict__ root1T,
                             float* __restrict__ xW1, float* __restrict__ rootout1) {
    int lane = threadIdx.x & 31;
    int tile = blockIdx.x * 8 + (threadIdx.x >> 5);
    if (tile >= NODES / 16) return;
    int r = blockIdx.y;
    const __bf16* BT = (r < N_REL) ? (W1T + (size_t)r * HID_DIM * IN_DIM) : root1T;
    int row0 = tile * 16;
    int arow = row0 + (lane & 15);
    v8f acc0 = {}, acc1 = {};
#pragma unroll
    for (int kk = 0; kk < IN_DIM; kk += 32) {
        v16bf a  = load_a_frag_bf(xb, IN_DIM, arow, kk, lane);
        v16bf b0 = load_b_frag_bf(BT, IN_DIM, (lane & 15), kk, lane);
        v16bf b1 = load_b_frag_bf(BT, IN_DIM, 16 + (lane & 15), kk, lane);
        acc0 = wmma_bf16(a, b0, acc0);
        acc1 = wmma_bf16(a, b1, acc1);
    }
    if (r < N_REL) {  // [N, R, HID] layout, row stride R*HID
        float* D = xW1 + (size_t)r * HID_DIM;
        store_tile(D, N_REL * HID_DIM, row0, 0, lane, acc0);
        store_tile(D, N_REL * HID_DIM, row0, 16, lane, acc1);
    } else {
        store_tile(rootout1, HID_DIM, row0, 0, lane, acc0);
        store_tile(rootout1, HID_DIM, row0, 16, lane, acc1);
    }
}

// One wave per edge, lane = hidden channel: agg1[dst,:] += xW1[src,etype,:]
__global__ void k_edge1(const int* __restrict__ src, const int* __restrict__ dst,
                        const int* __restrict__ etype,
                        const float* __restrict__ xW1,
                        float* __restrict__ agg1, float* __restrict__ deg) {
    long long tid = (long long)blockIdx.x * blockDim.x + threadIdx.x;
    int e = (int)(tid >> 5);
    int lane = (int)(tid & 31);
    if (e >= EDGES) return;
    if (lane == 0 && e + 8192 < EDGES) {   // sequential index streams -> HW prefetch
        __builtin_prefetch(&src[e + 8192], 0, 1);
        __builtin_prefetch(&dst[e + 8192], 0, 1);
        __builtin_prefetch(&etype[e + 8192], 0, 1);
    }
    int s = src[e], d = dst[e], t = etype[e];
    float m = xW1[((size_t)s * N_REL + t) * HID_DIM + lane];
    atomicAdd(&agg1[(size_t)d * HID_DIM + lane], m);
    if (lane == 0) atomicAdd(&deg[d], 1.0f);
}

// h1 = relu(agg1 / max(deg,1) + rootout1 + b1)
__global__ void k_final1(const float* __restrict__ agg1, const float* __restrict__ deg,
                         const float* __restrict__ rootout1, const float* __restrict__ b1,
                         float* __restrict__ h1) {
    long long tid = (long long)blockIdx.x * blockDim.x + threadIdx.x;
    if (tid >= (long long)NODES * HID_DIM) return;
    int n = (int)(tid / HID_DIM), j = (int)(tid % HID_DIM);
    float dg = deg[n];
    dg = dg > 1.0f ? dg : 1.0f;
    float v = agg1[tid] / dg + rootout1[tid] + b1[j];
    h1[tid] = v > 0.0f ? v : 0.0f;
}

// ---- Layer 2 ----------------------------------------------------------------

// Aggregate-first: s2[dst, etype, :] += h1[src, :]
__global__ void k_edge2(const int* __restrict__ src, const int* __restrict__ dst,
                        const int* __restrict__ etype,
                        const float* __restrict__ h1, float* __restrict__ s2) {
    long long tid = (long long)blockIdx.x * blockDim.x + threadIdx.x;
    int e = (int)(tid >> 5);
    int lane = (int)(tid & 31);
    if (e >= EDGES) return;
    if (lane == 0 && e + 8192 < EDGES) {
        __builtin_prefetch(&src[e + 8192], 0, 1);
        __builtin_prefetch(&dst[e + 8192], 0, 1);
        __builtin_prefetch(&etype[e + 8192], 0, 1);
    }
    int s = src[e], d = dst[e], t = etype[e];
    atomicAdd(&s2[((size_t)d * N_REL + t) * HID_DIM + lane],
              h1[(size_t)s * HID_DIM + lane]);
}

// h2 = (sum_r s2[:,r,:] @ W2[r]) / max(deg,1) + h1 @ root2 + b2
__global__ void k_layer2(const float* __restrict__ s2, const float* __restrict__ h1,
                         const float* __restrict__ deg,
                         const __bf16* __restrict__ W2T, const __bf16* __restrict__ root2T,
                         const float* __restrict__ b2, float* __restrict__ h2) {
    int lane = threadIdx.x & 31;
    int tile = blockIdx.x * 8 + (threadIdx.x >> 5);
    if (tile >= NODES / 16) return;
    int row0 = tile * 16;
    int arow = row0 + (lane & 15);
    v8f acc[4] = {{}, {}, {}, {}};
#pragma unroll
    for (int r = 0; r < N_REL; ++r) {
        v16bf a = load_a_frag_f32(s2 + (size_t)r * HID_DIM, N_REL * HID_DIM, arow, 0, lane);
        const __bf16* BT = W2T + (size_t)r * OUT_DIM * HID_DIM;
#pragma unroll
        for (int ct = 0; ct < 4; ++ct) {
            v16bf b = load_b_frag_bf(BT, HID_DIM, ct * 16 + (lane & 15), 0, lane);
            acc[ct] = wmma_bf16(a, b, acc[ct]);
        }
    }
    // per-row mean scale (row M = v + (lane>=16?8:0))
    int g = (lane & 16) ? 8 : 0;
    float inv[8];
#pragma unroll
    for (int v = 0; v < 8; ++v) {
        float dg = deg[row0 + g + v];
        inv[v] = 1.0f / (dg > 1.0f ? dg : 1.0f);
    }
#pragma unroll
    for (int ct = 0; ct < 4; ++ct)
#pragma unroll
        for (int v = 0; v < 8; ++v) acc[ct][v] *= inv[v];
    // root transform accumulates onto the scaled aggregate
    v16bf ar = load_a_frag_f32(h1, HID_DIM, arow, 0, lane);
#pragma unroll
    for (int ct = 0; ct < 4; ++ct) {
        v16bf b = load_b_frag_bf(root2T, HID_DIM, ct * 16 + (lane & 15), 0, lane);
        acc[ct] = wmma_bf16(ar, b, acc[ct]);
    }
#pragma unroll
    for (int ct = 0; ct < 4; ++ct) {
        float bb = b2[ct * 16 + (lane & 15)];
#pragma unroll
        for (int v = 0; v < 8; ++v) acc[ct][v] += bb;
        store_tile(h2, OUT_DIM, row0, ct * 16, lane, acc[ct]);
    }
}

// Final linear head: out[n, 0:2] = h2[n,:] @ lin_w + lin_b  (tiny, VALU)
__global__ void k_out(const float* __restrict__ h2, const float* __restrict__ lin_w,
                      const float* __restrict__ lin_b, float* __restrict__ out) {
    int n = blockIdx.x * blockDim.x + threadIdx.x;
    if (n >= NODES) return;
    float s0 = lin_b[0], s1 = lin_b[1];
    const float* h = h2 + (size_t)n * OUT_DIM;
#pragma unroll 8
    for (int j = 0; j < OUT_DIM; ++j) {
        float v = h[j];
        s0 += v * lin_w[j * 2 + 0];
        s1 += v * lin_w[j * 2 + 1];
    }
    out[n * 2 + 0] = s0;
    out[n * 2 + 1] = s1;
}

// ---- Launch ----------------------------------------------------------------
extern "C" void kernel_launch(void* const* d_in, const int* in_sizes, int n_in,
                              void* d_out, int out_size, void* d_ws, size_t ws_size,
                              hipStream_t stream) {
    const float* x      = (const float*)d_in[0];
    const int*   eidx   = (const int*)d_in[1];   // [2, E]: src then dst
    const int*   etype  = (const int*)d_in[2];
    const float* W1     = (const float*)d_in[3];
    const float* root1  = (const float*)d_in[4];
    const float* b1     = (const float*)d_in[5];
    const float* W2     = (const float*)d_in[6];
    const float* root2  = (const float*)d_in[7];
    const float* b2     = (const float*)d_in[8];
    const float* lin_w  = (const float*)d_in[9];
    const float* lin_b  = (const float*)d_in[10];
    float* out = (float*)d_out;

    const int* src = eidx;
    const int* dst = eidx + EDGES;

    // workspace layout; s2 aliases xW1 (xW1 dead before s2 is zeroed)
    char* ws = (char*)d_ws;
    float* xW1      = (float*)ws;                               // 25,600,000 f
    float* agg1     = xW1 + (size_t)NODES * N_REL * HID_DIM;    //  3,200,000 f
    float* deg      = agg1 + (size_t)NODES * HID_DIM;           //    100,000 f
    float* h1       = deg + NODES;                              //  3,200,000 f
    float* rootout1 = h1 + (size_t)NODES * HID_DIM;             //  3,200,000 f
    float* h2       = rootout1 + (size_t)NODES * HID_DIM;       //  6,400,000 f
    __bf16* xb      = (__bf16*)(h2 + (size_t)NODES * OUT_DIM);  // 12,800,000 bf
    __bf16* W1T     = xb + (size_t)NODES * IN_DIM;              //     32,768 bf
    __bf16* root1T  = W1T + (size_t)N_REL * IN_DIM * HID_DIM;   //      4,096 bf
    __bf16* W2T     = root1T + (size_t)IN_DIM * HID_DIM;        //     16,384 bf
    __bf16* root2T  = W2T + (size_t)N_REL * HID_DIM * OUT_DIM;  //      2,048 bf
    float* s2       = xW1;                                      // alias

    const int NTILES = NODES / 16;               // 6250
    const int TBLK = (NTILES + 7) / 8;           // 782 blocks, 8 waves each
    const int EBLK = (EDGES * 32 + 255) / 256;   // one wave per edge

    // precision/layout prep + zero agg1+deg (contiguous)
    k_cvt_x<<<(int)(((long long)NODES * IN_DIM / 8 + 255) / 256), 256, 0, stream>>>(x, xb);
    k_cvt_weights<<<(N_REL * IN_DIM * HID_DIM + 255) / 256, 256, 0, stream>>>(
        W1, root1, W2, root2, W1T, root1T, W2T, root2T);
    k_zero<<<2048, 256, 0, stream>>>(agg1, (long long)NODES * HID_DIM + NODES);
    // layer-1 dense transforms (8 relations + root) via WMMA
    k_transform1<<<dim3(TBLK, N_REL + 1), 256, 0, stream>>>(xb, W1T, root1T, xW1, rootout1);
    // layer-1 gather / scatter-mean
    k_edge1<<<EBLK, 256, 0, stream>>>(src, dst, etype, xW1, agg1, deg);
    k_final1<<<(int)(((long long)NODES * HID_DIM + 255) / 256), 256, 0, stream>>>(
        agg1, deg, rootout1, b1, h1);
    // layer-2: zero s2 (reusing xW1 storage), aggregate raw h1 per relation
    k_zero<<<4096, 256, 0, stream>>>(s2, (long long)NODES * N_REL * HID_DIM);
    k_edge2<<<EBLK, 256, 0, stream>>>(src, dst, etype, h1, s2);
    // layer-2 GEMM (per-relation + root) via WMMA, mean + bias fused
    k_layer2<<<TBLK, 256, 0, stream>>>(s2, h1, deg, W2T, root2T, b2, h2);
    // classifier head
    k_out<<<(NODES + 255) / 256, 256, 0, stream>>>(h2, lin_w, lin_b, out);
}
